// AttentionModel_67121748902395
// MI455X (gfx1250) — matrix operand order repaired and checked by reading, not compiled, not run on hardware
//
#include <hip/hip_runtime.h>

// Problem constants (match reference)
#define BATCH 4
#define CH    128
#define NPIX  4096   // 64*64

typedef __bf16 bf16_t;
typedef __attribute__((ext_vector_type(16))) bf16_t v16bf;
typedef __attribute__((ext_vector_type(8)))  float  v8f;
typedef __attribute__((ext_vector_type(8)))  int    v8i;
typedef __attribute__((ext_vector_type(4)))  int    v4i;

// Async global->LDS (CDNA5): use if this toolchain exposes the builtin.
#if defined(__has_builtin)
#  if __has_builtin(__builtin_amdgcn_global_load_async_to_lds_b128)
#    define HAVE_ASYNC_LDS 1
#  endif
#endif
#ifndef HAVE_ASYNC_LDS
#  define HAVE_ASYNC_LDS 0
#endif

// Probe-derived signature: (v4i AS1* src, v4i AS3* dst, imm offset, imm cpol)
typedef v4i __attribute__((address_space(1)))* gptr_v4i;
typedef v4i __attribute__((address_space(3)))* lptr_v4i;

__device__ __forceinline__ void wait_asynccnt0() {
#if defined(__has_builtin)
#  if __has_builtin(__builtin_amdgcn_s_wait_asynccnt)
    __builtin_amdgcn_s_wait_asynccnt(0);
    return;
#  endif
#endif
    asm volatile("s_wait_asynccnt 0x0" ::: "memory");
}

__device__ __forceinline__ v8f zero8() {
    v8f z = {0.f,0.f,0.f,0.f,0.f,0.f,0.f,0.f};
    return z;
}

// A-matrix (16x32, 16-bit) K index for VGPR slot v, lane-half h (CDNA5 ISA 7.12.2)
__device__ __forceinline__ int kmapA(int v, int h) {
    return (v < 4) ? (8*h + 2*v) : (16 + 8*h + 2*(v-4));
}
// B-matrix (32x16, 16-bit) K index for VGPR slot v, lane-half h
__device__ __forceinline__ int kmapB(int v, int h) {
    return (v < 4) ? (8*h + 2*v) : (16 + 8*h + 2*(v-4));
}

// ---------------------------------------------------------------------------
// Kernel 1: QKV projection (1x1 conv == GEMM over channels).
//   D[o, n] = sum_c W[o,c] * x[b,c,n]   (+bias, Q additionally * 1/sqrt(C))
// Stored as bf16 [B][N][C] row-major (row = pixel n, col = channel o).
// grid = (N/16, B, 3), block = 128 (4 waves); wave w -> o-tiles {32w, 32w+16}
// ---------------------------------------------------------------------------
__global__ __launch_bounds__(128)
void proj_qkv_kernel(const float* __restrict__ x,
                     const float* __restrict__ Wq, const float* __restrict__ bq,
                     const float* __restrict__ Wk, const float* __restrict__ bk,
                     const float* __restrict__ Wv, const float* __restrict__ bv,
                     bf16_t* __restrict__ Qb, bf16_t* __restrict__ Kb,
                     bf16_t* __restrict__ Vb)
{
    const int n0 = blockIdx.x * 16;
    const int b  = blockIdx.y;
    const int z  = blockIdx.z;

    const float* Wm; const float* bias; bf16_t* outp; float scale;
    if (z == 0)      { Wm = Wq; bias = bq; outp = Qb; scale = rsqrtf((float)CH); }
    else if (z == 1) { Wm = Wk; bias = bk; outp = Kb; scale = 1.0f; }
    else             { Wm = Wv; bias = bv; outp = Vb; scale = 1.0f; }

    __shared__ float xs[CH][20];   // x tile [c][n], padded row stride (bank-safe)

    const int tid = threadIdx.x;
    {   // stage: thread t loads channel row c=t (16 contiguous floats)
        const float* src = x + ((size_t)b * CH + tid) * NPIX + n0;
        float4 a0 = *(const float4*)(src + 0);
        float4 a1 = *(const float4*)(src + 4);
        float4 a2 = *(const float4*)(src + 8);
        float4 a3 = *(const float4*)(src + 12);
        float* dst = xs[tid];
        *(float4*)(dst + 0)  = a0;
        *(float4*)(dst + 4)  = a1;
        *(float4*)(dst + 8)  = a2;
        *(float4*)(dst + 12) = a3;
    }
    __syncthreads();

    const int wave = tid >> 5;
    const int lane = tid & 31;
    const int nl   = lane & 15;
    const int h    = lane >> 4;
    const int o0   = wave * 32;

    v8f acc[2] = { zero8(), zero8() };

    #pragma unroll
    for (int kb = 0; kb < 4; ++kb) {
        const int k0 = kb * 32;
        // B fragment: x tile (32c x 16n); column n = nl, K = channel
        v16bf bfrag;
        #pragma unroll
        for (int v = 0; v < 8; ++v) {
            int k = k0 + kmapB(v, h);
            bfrag[2*v]   = (bf16_t)xs[k][nl];
            bfrag[2*v+1] = (bf16_t)xs[k+1][nl];
        }
        #pragma unroll
        for (int t = 0; t < 2; ++t) {
            // A fragment: W tile (16o x 32c); row o = o0 + 16t + nl
            const float* wr = Wm + (size_t)(o0 + t*16 + nl) * CH + k0;
            v16bf afrag;
            #pragma unroll
            for (int v = 0; v < 8; ++v) {
                int kl = kmapA(v, h);
                float2 p = *(const float2*)(wr + kl);
                afrag[2*v]   = (bf16_t)p.x;
                afrag[2*v+1] = (bf16_t)p.y;
            }
            acc[t] = __builtin_amdgcn_wmma_f32_16x16x32_bf16(
                         false, afrag, false, bfrag, (short)0, acc[t],
                         false, false);
        }
    }

    // D layout: element (M=o_l, N=n_l): lane = n_l + 16*(o_l>=8), vgpr r = o_l%8
    bf16_t* op = outp + (size_t)b * NPIX * CH;
    #pragma unroll
    for (int t = 0; t < 2; ++t) {
        #pragma unroll
        for (int r = 0; r < 8; ++r) {
            int o = o0 + t*16 + r + 8*h;
            float val = (acc[t][r] + bias[o]) * scale;
            op[(size_t)(n0 + nl) * CH + o] = (bf16_t)val;
        }
    }
}

// ---------------------------------------------------------------------------
// Kernel 2: flash attention over N=4096 keys, 32 keys/iteration.
// grid = (N/128, B), block = 256 (8 waves); wave w -> 16 query rows.
// ---------------------------------------------------------------------------
__global__ __launch_bounds__(256)
void flash_attn_kernel(const bf16_t* __restrict__ Qb,
                       const bf16_t* __restrict__ Kb,
                       const bf16_t* __restrict__ Vb,
                       const float*  __restrict__ x,
                       const float*  __restrict__ gamma,
                       float* __restrict__ out)
{
    const int b    = blockIdx.y;
    const int tid  = threadIdx.x;
    const int wave = tid >> 5;
    const int lane = tid & 31;
    const int nl   = lane & 15;
    const int h    = lane >> 4;
    const int n0   = blockIdx.x * 128 + wave * 16;

    __shared__ bf16_t Ks[32][CH + 8];    // key tile   [m][c], padded
    __shared__ bf16_t Vt[CH][32 + 8];    // value tile [c][m] (transposed), padded
    __shared__ bf16_t Ps[8][16][34];     // per-wave P tile [n][m], padded
    __shared__ float  Os[8][16][20];     // per-wave O transpose scratch [o][n]

    // ---- load Q fragments (rows n0..n0+15, all 128 channels) ----
    const bf16_t* qbase = Qb + (size_t)b * NPIX * CH + (size_t)(n0 + nl) * CH;
    v16bf qf[4];
    #pragma unroll
    for (int kb = 0; kb < 4; ++kb) {
        v8i bits;
        #pragma unroll
        for (int v = 0; v < 8; ++v) {
            int kl = kmapA(v, h);
            bits[v] = *(const int*)(qbase + kb*32 + kl);
        }
        qf[kb] = __builtin_bit_cast(v16bf, bits);
    }

    v8f O[8];
    #pragma unroll
    for (int t = 0; t < 8; ++t) O[t] = zero8();
    float M[8], L[8];
    #pragma unroll
    for (int r = 0; r < 8; ++r) { M[r] = -INFINITY; L[r] = 0.f; }

    const bf16_t* Kbase = Kb + (size_t)b * NPIX * CH;
    const bf16_t* Vbase = Vb + (size_t)b * NPIX * CH;

    for (int m0 = 0; m0 < NPIX; m0 += 32) {
        // ---- stage K tile [32][128] (async global->LDS when available) ----
        {
            int row = tid >> 3;
            int c0  = (tid & 7) * 16;
            const bf16_t* src = Kbase + (size_t)(m0 + row) * CH + c0;
#if HAVE_ASYNC_LDS
            __builtin_amdgcn_global_load_async_to_lds_b128(
                (gptr_v4i)(src),
                (lptr_v4i)(&Ks[row][c0]), 0, 0);
            __builtin_amdgcn_global_load_async_to_lds_b128(
                (gptr_v4i)(src + 8),
                (lptr_v4i)(&Ks[row][c0 + 8]), 0, 0);
#else
            const uint4* s4 = (const uint4*)src;
            uint4 d0 = s4[0], d1 = s4[1];
            *(uint4*)(&Ks[row][c0])     = d0;
            *(uint4*)(&Ks[row][c0 + 8]) = d1;
#endif
        }
        // ---- stage V tile transposed -> Vt[c][m] ----
        {
            int row = tid >> 3;
            int c0  = (tid & 7) * 16;
            const bf16_t* src = Vbase + (size_t)(m0 + row) * CH + c0;
            union { uint4 u; bf16_t e[8]; } a0, a1;
            a0.u = *(const uint4*)(src);
            a1.u = *(const uint4*)(src + 8);
            #pragma unroll
            for (int j = 0; j < 8; ++j) {
                Vt[c0 + j][row]     = a0.e[j];
                Vt[c0 + 8 + j][row] = a1.e[j];
            }
        }
#if HAVE_ASYNC_LDS
        wait_asynccnt0();
#endif
        __syncthreads();

        // ---- S = Q * K^T  (two 16x16 tiles, K-dim = 128) ----
        v8f S0 = zero8(), S1 = zero8();
        #pragma unroll
        for (int kb = 0; kb < 4; ++kb) {
            v8i b0, b1;
            #pragma unroll
            for (int v = 0; v < 8; ++v) {
                int k = kb*32 + kmapB(v, h);
                b0[v] = *(const int*)(&Ks[nl][k]);
                b1[v] = *(const int*)(&Ks[nl + 16][k]);
            }
            S0 = __builtin_amdgcn_wmma_f32_16x16x32_bf16(
                     false, qf[kb], false, __builtin_bit_cast(v16bf, b0),
                     (short)0, S0, false, false);
            S1 = __builtin_amdgcn_wmma_f32_16x16x32_bf16(
                     false, qf[kb], false, __builtin_bit_cast(v16bf, b1),
                     (short)0, S1, false, false);
        }

        // ---- online softmax (row = r + 8h lives in 16-lane half h) ----
        float alpha[8];
        #pragma unroll
        for (int r = 0; r < 8; ++r) {
            float t = fmaxf(S0[r], S1[r]);
            #pragma unroll
            for (int d = 1; d < 16; d <<= 1) t = fmaxf(t, __shfl_xor(t, d, 16));
            float mn = fmaxf(M[r], t);
            float a  = __expf(M[r] - mn);
            float p0 = __expf(S0[r] - mn);
            float p1 = __expf(S1[r] - mn);
            float s  = p0 + p1;
            #pragma unroll
            for (int d = 1; d < 16; d <<= 1) s += __shfl_xor(s, d, 16);
            L[r] = L[r] * a + s;
            M[r] = mn;
            alpha[r] = a;
            // stash P in LDS (D layout -> [n][m] tile for A-frag rebuild)
            Ps[wave][r + 8*h][nl]      = (bf16_t)p0;
            Ps[wave][r + 8*h][nl + 16] = (bf16_t)p1;
        }
        asm volatile("s_wait_dscnt 0x0" ::: "memory");   // wave-local LDS RAW

        // ---- rebuild P as A fragment (16n x 32m) ----
        v16bf pf;
        {
            v8i bits;
            #pragma unroll
            for (int v = 0; v < 8; ++v) {
                int kl = kmapA(v, h);
                bits[v] = *(const int*)(&Ps[wave][nl][kl]);
            }
            pf = __builtin_bit_cast(v16bf, bits);
        }

        // ---- O = O*alpha + P * V  (8 o-tiles, K-dim = 32) ----
        #pragma unroll
        for (int t = 0; t < 8; ++t) {
            #pragma unroll
            for (int r = 0; r < 8; ++r) O[t][r] *= alpha[r];
            v8i bits;
            #pragma unroll
            for (int v = 0; v < 8; ++v) {
                int k = kmapB(v, h);
                bits[v] = *(const int*)(&Vt[t*16 + nl][k]);
            }
            O[t] = __builtin_amdgcn_wmma_f32_16x16x32_bf16(
                       false, pf, false, __builtin_bit_cast(v16bf, bits),
                       (short)0, O[t], false, false);
        }
        __syncthreads();   // before K/V tiles are overwritten
    }

    // ---- epilogue: out[b][o][n] = gamma * O/L + x, coalesced along n ----
    // Per tile: transpose O through per-wave LDS so lane pairs handle one
    // o-row with float4 accesses over contiguous n.
    const float g   = gamma[0];
    const float* xb = x   + (size_t)b * CH * NPIX;
    float*       ob = out + (size_t)b * CH * NPIX;
    float rL[8];
    #pragma unroll
    for (int r = 0; r < 8; ++r) rL[r] = 1.0f / L[r];

    const int orow = lane >> 1;          // 0..15: o within tile
    const int part = lane & 1;           // two lanes split the 16 n's
    #pragma unroll
    for (int t = 0; t < 8; ++t) {
        // scatter tile into [o][n] scratch (normalized)
        #pragma unroll
        for (int r = 0; r < 8; ++r)
            Os[wave][nl][r + 8*h] = O[t][r] * rL[r];
        asm volatile("s_wait_dscnt 0x0" ::: "memory");   // wave-local LDS RAW

        const float* srow = Os[wave][orow] + part * 8;
        float4 v0 = *(const float4*)(srow + 0);
        float4 v1 = *(const float4*)(srow + 4);

        size_t base = (size_t)(t*16 + orow) * NPIX + n0 + part * 8;
        float4 x0 = *(const float4*)(xb + base);
        float4 x1 = *(const float4*)(xb + base + 4);
        float4 r0, r1;
        r0.x = g*v0.x + x0.x; r0.y = g*v0.y + x0.y;
        r0.z = g*v0.z + x0.z; r0.w = g*v0.w + x0.w;
        r1.x = g*v1.x + x1.x; r1.y = g*v1.y + x1.y;
        r1.z = g*v1.z + x1.z; r1.w = g*v1.w + x1.w;
        *(float4*)(ob + base)     = r0;
        *(float4*)(ob + base + 4) = r1;
        asm volatile("s_wait_dscnt 0x0" ::: "memory");   // before scratch reuse
    }
}

// ---------------------------------------------------------------------------
extern "C" void kernel_launch(void* const* d_in, const int* in_sizes, int n_in,
                              void* d_out, int out_size, void* d_ws, size_t ws_size,
                              hipStream_t stream) {
    const float* x     = (const float*)d_in[0];
    const float* Wq    = (const float*)d_in[1];
    const float* bq    = (const float*)d_in[2];
    const float* Wk    = (const float*)d_in[3];
    const float* bk    = (const float*)d_in[4];
    const float* Wv    = (const float*)d_in[5];
    const float* bv    = (const float*)d_in[6];
    const float* gamma = (const float*)d_in[7];

    const size_t per = (size_t)BATCH * NPIX * CH;   // elements per bf16 buffer
    bf16_t* Qb = (bf16_t*)d_ws;
    bf16_t* Kb = Qb + per;
    bf16_t* Vb = Kb + per;

    dim3 g1(NPIX / 16, BATCH, 3);
    proj_qkv_kernel<<<g1, 128, 0, stream>>>(x, Wq, bq, Wk, bk, Wv, bv, Qb, Kb, Vb);

    dim3 g2(NPIX / 128, BATCH);
    flash_attn_kernel<<<g2, 256, 0, stream>>>(Qb, Kb, Vb, x, gamma, (float*)d_out);
}
